// MixTransformer_24300924961063
// MI455X (gfx1250) — compile-verified, matrix operand order checked by reading, not run on hardware
//
#include <hip/hip_runtime.h>
#include <stdint.h>

typedef __bf16 bf16;
typedef __attribute__((ext_vector_type(16))) __bf16 bf16x16;
typedef __attribute__((ext_vector_type(8)))  float  f32x8;

#define DI __device__ __forceinline__

// ---- model dims (compile-time) ----
constexpr int Bc   = 2;
constexpr int Sc   = 1024;
constexpr int Dc   = 2048;
constexpr int Hc   = 32;
constexpr int KVHc = 8;
constexpr int HDc  = 64;
constexpr int NREPc = Hc / KVHc;
constexpr int Fc   = 5632;
constexpr int Ec   = 8;
constexpr int Rc   = 16;
constexpr int Mc   = Bc * Sc;        // 2048 tokens
constexpr int KVc  = KVHc * HDc;     // 512
constexpr int LDVc = Sc + 16;        // padded vt row length
constexpr float kLoraScale = 2.0f;   // 32/16

// ---------------------------------------------------------------------------
// WMMA helpers (CDNA5 bf16 16x16x32, fp32 accumulate)
// ---------------------------------------------------------------------------
DI f32x8 wmma_bf16(bf16x16 a, bf16x16 b, f32x8 c) {
  return __builtin_amdgcn_wmma_f32_16x16x32_bf16(false, a, false, b, (short)0, c,
                                                 false, false);
}

// Load a 16x32 bf16 fragment (A layout; also B^T rows) from row-major memory.
// lane 0-15 : row = row0+lane,    K = k0+{0..7, 16..23}
// lane 16-31: row = row0+lane-16, K = k0+{8..15,24..31}
DI bf16x16 ld_frag(const bf16* __restrict__ p, int ld, int row0, int k0, int lane) {
  const bf16* q = p + (size_t)(row0 + (lane & 15)) * ld + (k0 + ((lane >> 4) << 3));
  union { bf16x16 v; uint4 u[2]; } t;
  t.u[0] = *(const uint4*)q;        // K kb..kb+7   -> vgpr 0-3
  t.u[1] = *(const uint4*)(q + 16); // K kb+16..+23 -> vgpr 4-7
  return t.v;
}

// Same fragment read from an LDS tile with fixed row stride of 32 bf16.
DI bf16x16 ld_frag_lds(const bf16* base, int row0, int lane) {
  const bf16* q = base + (row0 + (lane & 15)) * 32 + ((lane >> 4) << 3);
  union { bf16x16 v; uint4 u[2]; } t;
  t.u[0] = *(const uint4*)q;
  t.u[1] = *(const uint4*)(q + 16);
  return t.v;
}

// Async copy of one 16-byte chunk global -> LDS (gfx1250, ASYNCcnt tracked).
DI void async_ld_b128(const void* gaddr, void* lds_ptr) {
  const uint64_t ga = (uint64_t)(uintptr_t)gaddr;
  const uint32_t la = (uint32_t)(uintptr_t)lds_ptr;  // LDS offset = addr[31:0]
  asm volatile("global_load_async_to_lds_b128 %0, %1, off" ::"v"(la), "v"(ga)
               : "memory");
}
DI void wait_async0() { asm volatile("s_wait_asynccnt 0x0" ::: "memory"); }

// ---------------------------------------------------------------------------
// Generic GEMM: C[M,N] = A[M,K] * B[N,K]^T   (A,B bf16 row-major, K%32==0,
// M%128==0, N%128==0). Stages both 128x32 tiles in LDS via async copies.
// Block: 256 thr = 8 waves (4 m-waves x 2 n-waves); wave tile 32x64.
// ---------------------------------------------------------------------------
template <bool WF, bool WB>
__global__ __launch_bounds__(256) void gemm_bf16_k(const bf16* __restrict__ A,
                                                   const bf16* __restrict__ Bm,
                                                   float* __restrict__ Cf,
                                                   bf16* __restrict__ Cb,
                                                   int N, int K) {
  __shared__ alignas(16) bf16 at[128][32];
  __shared__ alignas(16) bf16 bt[128][32];
  const int tid = threadIdx.x;
  const int lane = tid & 31;
  const int wave = tid >> 5;
  const int wm = wave & 3;
  const int wn = wave >> 2;
  const int mb = blockIdx.y * 128;
  const int nb = blockIdx.x * 128;
  const int m0 = wm * 32;  // within block tile
  const int n0 = wn * 64;

  f32x8 acc[2][4] = {};

  for (int k0 = 0; k0 < K; k0 += 32) {
    // Stage A and B 128x32 tiles: 512 16B chunks each, 2 per thread per tile.
#pragma unroll
    for (int u = 0; u < 2; ++u) {
      const int c = tid * 2 + u;
      const int row = c >> 2;       // 0..127
      const int seg = (c & 3) * 8;  // bf16 offset of 16B chunk
      async_ld_b128(A + (size_t)(mb + row) * K + (k0 + seg), &at[row][seg]);
      async_ld_b128(Bm + (size_t)(nb + row) * K + (k0 + seg), &bt[row][seg]);
    }
    wait_async0();
    __syncthreads();

    bf16x16 a[2], b[4];
#pragma unroll
    for (int i = 0; i < 2; ++i) a[i] = ld_frag_lds(&at[0][0], m0 + 16 * i, lane);
#pragma unroll
    for (int j = 0; j < 4; ++j) b[j] = ld_frag_lds(&bt[0][0], n0 + 16 * j, lane);
#pragma unroll
    for (int i = 0; i < 2; ++i)
#pragma unroll
      for (int j = 0; j < 4; ++j) acc[i][j] = wmma_bf16(a[i], b[j], acc[i][j]);
    __syncthreads();
  }

  // Branch-free epilogue (C layout: row = v + 8*half, col = lane&15 per tile).
  const int half = lane >> 4, nn = lane & 15;
#pragma unroll
  for (int i = 0; i < 2; ++i) {
    const int grow0 = mb + m0 + 16 * i + 8 * half;
#pragma unroll
    for (int v = 0; v < 8; ++v) {
      const size_t rb = (size_t)(grow0 + v) * N + (nb + n0 + nn);
#pragma unroll
      for (int j = 0; j < 4; ++j) {
        const float val = acc[i][j][v];
        if constexpr (WF) Cf[rb + 16 * j] = val;
        if constexpr (WB) Cb[rb + 16 * j] = (bf16)val;
      }
    }
  }
}

// ---------------------------------------------------------------------------
// Small-N GEMM: C[M,16] = A[M,K] * B[16,K]^T (LoRA down-projection).
// Block: 256 thr = 8 waves; wave w handles rows blockIdx.x*128 + 16w.
// ---------------------------------------------------------------------------
__global__ __launch_bounds__(256) void gemm_bf16_n16(const bf16* __restrict__ A,
                                                     const bf16* __restrict__ Bm,
                                                     float* __restrict__ Cf,
                                                     int K) {
  const int lane = threadIdx.x & 31;
  const int wave = threadIdx.x >> 5;
  const int m0 = blockIdx.x * 128 + wave * 16;

  f32x8 acc = {};
  for (int k0 = 0; k0 < K; k0 += 32) {
    __builtin_prefetch(A + (size_t)(m0 + (lane & 15)) * K + k0 + 64, 0, 0);
    bf16x16 a = ld_frag(A, K, m0, k0, lane);
    bf16x16 b = ld_frag(Bm, K, 0, k0, lane);
    acc = wmma_bf16(a, b, acc);
  }
  const int half = lane >> 4, nn = lane & 15;
#pragma unroll
  for (int v = 0; v < 8; ++v)
    Cf[(size_t)(m0 + v + 8 * half) * Rc + nn] = acc[v];
}

// ---------------------------------------------------------------------------
// Flash attention: one wave per (b, h, 16-query tile). Causal, GQA.
// q: [B,S,H*HD] bf16 (post-rope), k: [B,S,KVH*HD] bf16, vt: [B,KVH,HD,LDV] bf16
// out: [B,S,H*HD] bf16
// ---------------------------------------------------------------------------
__global__ __launch_bounds__(32) void attn_kernel(const bf16* __restrict__ q,
                                                  const bf16* __restrict__ k,
                                                  const bf16* __restrict__ vt,
                                                  bf16* __restrict__ out) {
  const int lane = threadIdx.x;
  const int qt = blockIdx.x;
  const int bh = blockIdx.y;
  const int b = bh / Hc, h = bh % Hc;
  const int kvh = h / NREPc;
  const int q0 = qt * 16;
  const int ldq = Dc, ldk = KVc;

  const bf16* qbase = q + (size_t)b * Sc * ldq + h * HDc;
  const bf16* kbase = k + (size_t)b * Sc * ldk + kvh * HDc;
  const bf16* vbase = vt + ((size_t)b * KVHc + kvh) * HDc * LDVc;

  bf16x16 qa[2];
#pragma unroll
  for (int t = 0; t < 2; ++t) qa[t] = ld_frag(qbase, ldq, q0, 32 * t, lane);

  f32x8 o[4] = {};
  float rmax[8], rsum[8];
#pragma unroll
  for (int v = 0; v < 8; ++v) { rmax[v] = -1e30f; rsum[v] = 0.f; }

  __shared__ alignas(16) bf16 pl[16][40];  // 16x32 P tile, padded rows
  const int half = lane >> 4, nn = lane & 15;
  const int nkt = qt + 1;  // key tiles (causal)

  for (int kt = 0; kt < nkt; kt += 2) {
    f32x8 s0 = {}, s1 = {};
#pragma unroll
    for (int t = 0; t < 2; ++t) {
      bf16x16 kb = ld_frag(kbase, ldk, kt * 16, 32 * t, lane);
      s0 = wmma_bf16(qa[t], kb, s0);
    }
    const bool have1 = (kt + 1) < nkt;  // block-uniform
    if (have1) {
#pragma unroll
      for (int t = 0; t < 2; ++t) {
        bf16x16 kb = ld_frag(kbase, ldk, (kt + 1) * 16, 32 * t, lane);
        s1 = wmma_bf16(qa[t], kb, s1);
      }
    }
    // scale + causal mask (C layout: row m = v+8*half, col n = lane&15)
    float p0[8], p1[8], mx[8];
#pragma unroll
    for (int v = 0; v < 8; ++v) {
      const int m = q0 + v + 8 * half;
      const int kn = kt * 16 + nn;
      p0[v] = (kn <= m) ? s0[v] * 0.125f : -1e30f;
      p1[v] = (have1 && (kn + 16) <= m) ? s1[v] * 0.125f : -1e30f;
      mx[v] = fmaxf(p0[v], p1[v]);
    }
    // row max across the 16 lanes of each half-wave
#pragma unroll
    for (int off = 1; off < 16; off <<= 1)
#pragma unroll
      for (int v = 0; v < 8; ++v) mx[v] = fmaxf(mx[v], __shfl_xor(mx[v], off, 32));

    float psum[8];
#pragma unroll
    for (int v = 0; v < 8; ++v) {
      const float nm = fmaxf(rmax[v], mx[v]);
      const float alpha = __expf(rmax[v] - nm);
      rmax[v] = nm;
      rsum[v] *= alpha;
#pragma unroll
      for (int j = 0; j < 4; ++j) o[j][v] *= alpha;
      p0[v] = __expf(p0[v] - nm);
      p1[v] = __expf(p1[v] - nm);
      psum[v] = p0[v] + p1[v];
    }
#pragma unroll
    for (int off = 1; off < 16; off <<= 1)
#pragma unroll
      for (int v = 0; v < 8; ++v) psum[v] += __shfl_xor(psum[v], off, 32);
#pragma unroll
    for (int v = 0; v < 8; ++v) rsum[v] += psum[v];

    // transpose P (C layout -> A layout) through LDS
#pragma unroll
    for (int v = 0; v < 8; ++v) {
      pl[v + 8 * half][nn] = (bf16)p0[v];
      pl[v + 8 * half][nn + 16] = (bf16)p1[v];
    }
    __syncthreads();
    bf16x16 pa;
    {
      const bf16* pr = &pl[lane & 15][half * 8];
      union { bf16x16 v; uint4 u[2]; } t;
      t.u[0] = *(const uint4*)pr;
      t.u[1] = *(const uint4*)(pr + 16);
      pa = t.v;
    }
    __syncthreads();

    // P(16x32) @ V(32xHD): B^T rows are vt rows (contiguous keys)
#pragma unroll
    for (int j = 0; j < 4; ++j) {
      bf16x16 vb = ld_frag(vbase + (size_t)(j * 16) * LDVc, LDVc, 0, kt * 16, lane);
      o[j] = wmma_bf16(pa, vb, o[j]);
    }
  }

#pragma unroll
  for (int v = 0; v < 8; ++v) {
    const float inv = 1.0f / rsum[v];
#pragma unroll
    for (int j = 0; j < 4; ++j)
      out[(size_t)(b * Sc + q0 + v + 8 * half) * Dc + h * HDc + j * 16 + nn] =
          (bf16)(o[j][v] * inv);
  }
}

// ---------------------------------------------------------------------------
// Elementwise / reduction kernels
// ---------------------------------------------------------------------------
__global__ void f32_to_bf16_k(const float* __restrict__ in, bf16* __restrict__ out,
                              size_t n) {
  size_t i = (size_t)blockIdx.x * blockDim.x + threadIdx.x;
  const size_t stride = (size_t)gridDim.x * blockDim.x;
  for (; i < n; i += stride) out[i] = (bf16)in[i];
}

__global__ void zero_f32_k(float* __restrict__ x, size_t n) {
  size_t i = (size_t)blockIdx.x * blockDim.x + threadIdx.x;
  const size_t stride = (size_t)gridDim.x * blockDim.x;
  for (; i < n; i += stride) x[i] = 0.f;
}

__global__ __launch_bounds__(256) void rmsnorm_k(const float* __restrict__ x,
                                                 const float* __restrict__ w,
                                                 bf16* __restrict__ out) {
  const int row = blockIdx.x;
  const size_t off = (size_t)row * Dc;
  float s = 0.f;
  for (int i = threadIdx.x; i < Dc; i += 256) { float v = x[off + i]; s += v * v; }
#pragma unroll
  for (int o = 16; o; o >>= 1) s += __shfl_xor(s, o, 32);
  __shared__ float red[8];
  if ((threadIdx.x & 31) == 0) red[threadIdx.x >> 5] = s;
  __syncthreads();
  float tot = 0.f;
#pragma unroll
  for (int i = 0; i < 8; ++i) tot += red[i];
  const float inv = rsqrtf(tot / Dc + 1e-5f);
  for (int i = threadIdx.x; i < Dc; i += 256)
    out[off + i] = (bf16)(x[off + i] * inv * w[i]);
}

__global__ __launch_bounds__(256) void add_rmsnorm_k(const float* __restrict__ a,
                                                     const float* __restrict__ b,
                                                     const float* __restrict__ w,
                                                     float* __restrict__ resid,
                                                     bf16* __restrict__ out) {
  const int row = blockIdx.x;
  const size_t off = (size_t)row * Dc;
  float s = 0.f;
  for (int i = threadIdx.x; i < Dc; i += 256) {
    const float v = a[off + i] + b[off + i];
    resid[off + i] = v;
    s += v * v;
  }
#pragma unroll
  for (int o = 16; o; o >>= 1) s += __shfl_xor(s, o, 32);
  __shared__ float red[8];
  if ((threadIdx.x & 31) == 0) red[threadIdx.x >> 5] = s;
  __syncthreads();
  float tot = 0.f;
#pragma unroll
  for (int i = 0; i < 8; ++i) tot += red[i];
  const float inv = rsqrtf(tot / Dc + 1e-5f);
  for (int i = threadIdx.x; i < Dc; i += 256)
    out[off + i] = (bf16)(resid[off + i] * inv * w[i]);
}

// rope over fp32 [B,S,NH,HD] -> bf16, pair index layout matches memory
__global__ void rope_k(const float* __restrict__ x, const float* __restrict__ c,
                       const float* __restrict__ s, bf16* __restrict__ out, int NH,
                       int total_pairs) {
  const int idx = blockIdx.x * blockDim.x + threadIdx.x;
  if (idx >= total_pairs) return;
  const int hp = HDc / 2;
  const int pp = idx % hp;
  const int t = idx / hp;
  const int ss = (t / NH) % Sc;
  const float cv = c[ss * hp + pp], sv = s[ss * hp + pp];
  const float xr = x[2 * (size_t)idx], xi = x[2 * (size_t)idx + 1];
  out[2 * (size_t)idx] = (bf16)(xr * cv - xi * sv);
  out[2 * (size_t)idx + 1] = (bf16)(xr * sv + xi * cv);
}

// v [B,S,KVH,HD] -> vt [B,KVH,HD,LDV]
__global__ void transpose_v_k(const bf16* __restrict__ v, bf16* __restrict__ vt) {
  const int idx = blockIdx.x * blockDim.x + threadIdx.x;
  if (idx >= Bc * KVHc * HDc * Sc) return;
  const int s = idx % Sc;
  int r = idx / Sc;
  const int d = r % HDc; r /= HDc;
  const int h = r % KVHc;
  const int b = r / KVHc;
  vt[(((size_t)b * KVHc + h) * HDc + d) * LDVc + s] =
      v[(((size_t)b * Sc + s) * KVHc + h) * HDc + d];
}

// router: softmax over E=8 logits, top-2, renormalized weights
__global__ __launch_bounds__(256) void gate_k(const bf16* __restrict__ hn,
                                              const float* __restrict__ gw,
                                              float* __restrict__ ew) {
  const int token = blockIdx.x;
  const int e = threadIdx.x >> 5, lane = threadIdx.x & 31;
  const bf16* hr = hn + (size_t)token * Dc;
  const float* gr = gw + (size_t)e * Dc;
  float s = 0.f;
  for (int i = lane; i < Dc; i += 32) s += (float)hr[i] * gr[i];
#pragma unroll
  for (int o = 16; o; o >>= 1) s += __shfl_xor(s, o, 32);
  __shared__ float lg[Ec];
  if (lane == 0) lg[e] = s;
  __syncthreads();
  if (threadIdx.x == 0) {
    float mx = lg[0];
    for (int i = 1; i < Ec; ++i) mx = fmaxf(mx, lg[i]);
    float pr[Ec], den = 0.f;
    for (int i = 0; i < Ec; ++i) { pr[i] = __expf(lg[i] - mx); den += pr[i]; }
    for (int i = 0; i < Ec; ++i) pr[i] /= den;
    int i1 = 0;
    for (int i = 1; i < Ec; ++i) if (pr[i] > pr[i1]) i1 = i;
    int i2 = (i1 == 0) ? 1 : 0;
    for (int i = 0; i < Ec; ++i) if (i != i1 && pr[i] > pr[i2]) i2 = i;
    const float s2 = pr[i1] + pr[i2];
    for (int i = 0; i < Ec; ++i) ew[(size_t)token * Ec + i] = 0.f;
    ew[(size_t)token * Ec + i1] = pr[i1] / s2;
    ew[(size_t)token * Ec + i2] = pr[i2] / s2;
  }
}

// g = silu(c1 + scale*t1@B1^T) * (c3 + scale*t3@B3^T), rank-16 dot in VALU
__global__ void fused_lora_silu_k(const float* __restrict__ c1,
                                  const float* __restrict__ c3,
                                  const float* __restrict__ t1,
                                  const float* __restrict__ t3,
                                  const float* __restrict__ b1,
                                  const float* __restrict__ b3,
                                  bf16* __restrict__ g) {
  const size_t idx = (size_t)blockIdx.x * blockDim.x + threadIdx.x;
  if (idx >= (size_t)Mc * Fc) return;
  const int m = (int)(idx / Fc), f = (int)(idx % Fc);
  float d1 = 0.f, d3 = 0.f;
#pragma unroll
  for (int r = 0; r < Rc; ++r) {
    d1 += t1[m * Rc + r] * b1[(size_t)f * Rc + r];
    d3 += t3[m * Rc + r] * b3[(size_t)f * Rc + r];
  }
  const float v1 = c1[idx] + kLoraScale * d1;
  const float v3 = c3[idx] + kLoraScale * d3;
  const float sg = v1 / (1.f + __expf(-v1));
  g[idx] = (bf16)(sg * v3);
}

// final += ew[:,e] * (hs + scale * t2 @ B2^T)
__global__ void lora2_accum_k(const float* __restrict__ hs,
                              const float* __restrict__ t2,
                              const float* __restrict__ b2,
                              const float* __restrict__ ew, int e,
                              float* __restrict__ fin) {
  const size_t idx = (size_t)blockIdx.x * blockDim.x + threadIdx.x;
  if (idx >= (size_t)Mc * Dc) return;
  const int m = (int)(idx / Dc), d = (int)(idx % Dc);
  float dot = 0.f;
#pragma unroll
  for (int r = 0; r < Rc; ++r) dot += t2[m * Rc + r] * b2[(size_t)d * Rc + r];
  fin[idx] += ew[(size_t)m * Ec + e] * (hs[idx] + kLoraScale * dot);
}

__global__ void add_out_k(const float* __restrict__ a, const float* __restrict__ b,
                          float* __restrict__ o, size_t n) {
  size_t i = (size_t)blockIdx.x * blockDim.x + threadIdx.x;
  if (i < n) o[i] = a[i] + b[i];
}

// ---------------------------------------------------------------------------
// Host-side orchestration
// ---------------------------------------------------------------------------
extern "C" void kernel_launch(void* const* d_in, const int* in_sizes, int n_in,
                              void* d_out, int out_size, void* d_ws, size_t ws_size,
                              hipStream_t stream) {
  (void)in_sizes; (void)n_in; (void)out_size; (void)ws_size;
  const float* data  = (const float*)d_in[0];
  const float* cosb  = (const float*)d_in[2];
  const float* sinb  = (const float*)d_in[3];
  const float* anw   = (const float*)d_in[4];
  const float* fnw   = (const float*)d_in[5];
  const float* wq    = (const float*)d_in[6];
  const float* wk    = (const float*)d_in[7];
  const float* wv    = (const float*)d_in[8];
  const float* wo    = (const float*)d_in[9];
  const float* gatew = (const float*)d_in[10];
  const float* w1w   = (const float*)d_in[11];
  const float* w2w   = (const float*)d_in[12];
  const float* w3w   = (const float*)d_in[13];
  const float* w1A   = (const float*)d_in[14];
  const float* w1B   = (const float*)d_in[15];
  const float* w3A   = (const float*)d_in[16];
  const float* w3B   = (const float*)d_in[17];
  const float* w2A   = (const float*)d_in[18];
  const float* w2B   = (const float*)d_in[19];
  float* out = (float*)d_out;

  char* p = (char*)d_ws;
  auto alloc = [&](size_t bytes) -> char* {
    char* r = p;
    p += (bytes + 255) & ~(size_t)255;
    return r;
  };
  bf16* wq_b   = (bf16*)alloc((size_t)Dc * Dc * 2);
  bf16* wk_b   = (bf16*)alloc((size_t)KVc * Dc * 2);
  bf16* wv_b   = (bf16*)alloc((size_t)KVc * Dc * 2);
  bf16* wo_b   = (bf16*)alloc((size_t)Dc * Dc * 2);
  bf16* w1_b   = (bf16*)alloc((size_t)Fc * Dc * 2);
  bf16* w2_b   = (bf16*)alloc((size_t)Dc * Fc * 2);
  bf16* w3_b   = (bf16*)alloc((size_t)Fc * Dc * 2);
  bf16* w1A_b  = (bf16*)alloc((size_t)Ec * Rc * Fc * 2);
  bf16* w3A_b  = (bf16*)alloc((size_t)Ec * Rc * Fc * 2);
  bf16* w2A_b  = (bf16*)alloc((size_t)Ec * Rc * Dc * 2);
  bf16* h_b    = (bf16*)alloc((size_t)Mc * Dc * 2);
  float* qf    = (float*)alloc((size_t)Mc * Dc * 4);  // reused: q, wo-proj, hs
  float* kf    = (float*)alloc((size_t)Mc * KVc * 4);
  bf16* q_b    = (bf16*)alloc((size_t)Mc * Dc * 2);
  bf16* k_b    = (bf16*)alloc((size_t)Mc * KVc * 2);
  bf16* v_b    = (bf16*)alloc((size_t)Mc * KVc * 2);
  bf16* vt_b   = (bf16*)alloc((size_t)Bc * KVHc * HDc * LDVc * 2);
  bf16* attn_b = (bf16*)alloc((size_t)Mc * Dc * 2);
  float* resid = (float*)alloc((size_t)Mc * Dc * 4);
  bf16* hn_b   = (bf16*)alloc((size_t)Mc * Dc * 2);
  float* ew    = (float*)alloc((size_t)Mc * Ec * 4);
  float* c1f   = (float*)alloc((size_t)Mc * Fc * 4);
  float* c3f   = (float*)alloc((size_t)Mc * Fc * 4);
  bf16* c1b    = (bf16*)alloc((size_t)Mc * Fc * 2);
  bf16* c3b    = (bf16*)alloc((size_t)Mc * Fc * 2);
  float* t1f   = (float*)alloc((size_t)Mc * Rc * 4);
  float* t3f   = (float*)alloc((size_t)Mc * Rc * 4);
  bf16* g_b    = (bf16*)alloc((size_t)Mc * Fc * 2);
  bf16* hs_b   = (bf16*)alloc((size_t)Mc * Dc * 2);
  float* t2f   = (float*)alloc((size_t)Mc * Rc * 4);
  float* finf  = (float*)alloc((size_t)Mc * Dc * 4);

  auto cvt = [&](const float* s, bf16* d, size_t n) {
    f32_to_bf16_k<<<2048, 256, 0, stream>>>(s, d, n);
  };
  auto gemm = [&](const bf16* A, const bf16* Bw, float* Cf, bf16* Cb, int M, int N,
                  int K) {
    dim3 grid(N / 128, M / 128);
    if (Cf && Cb)
      gemm_bf16_k<true, true><<<grid, 256, 0, stream>>>(A, Bw, Cf, Cb, N, K);
    else if (Cf)
      gemm_bf16_k<true, false><<<grid, 256, 0, stream>>>(A, Bw, Cf, Cb, N, K);
    else
      gemm_bf16_k<false, true><<<grid, 256, 0, stream>>>(A, Bw, Cf, Cb, N, K);
  };
  auto gemm16 = [&](const bf16* A, const bf16* Bw, float* Cf, int K) {
    gemm_bf16_n16<<<Mc / 128, 256, 0, stream>>>(A, Bw, Cf, K);
  };

  // --- weight conversion (bf16) ---
  cvt(wq, wq_b, (size_t)Dc * Dc);
  cvt(wk, wk_b, (size_t)KVc * Dc);
  cvt(wv, wv_b, (size_t)KVc * Dc);
  cvt(wo, wo_b, (size_t)Dc * Dc);
  cvt(w1w, w1_b, (size_t)Fc * Dc);
  cvt(w2w, w2_b, (size_t)Dc * Fc);
  cvt(w3w, w3_b, (size_t)Fc * Dc);
  cvt(w1A, w1A_b, (size_t)Ec * Rc * Fc);
  cvt(w3A, w3A_b, (size_t)Ec * Rc * Fc);
  cvt(w2A, w2A_b, (size_t)Ec * Rc * Dc);

  // --- attention ---
  rmsnorm_k<<<Mc, 256, 0, stream>>>(data, anw, h_b);
  gemm(h_b, wq_b, qf, nullptr, Mc, Dc, Dc);
  gemm(h_b, wk_b, kf, nullptr, Mc, KVc, Dc);
  gemm(h_b, wv_b, nullptr, v_b, Mc, KVc, Dc);
  {
    const int qpairs = Mc * Dc / 2;
    rope_k<<<(qpairs + 255) / 256, 256, 0, stream>>>(qf, cosb, sinb, q_b, Hc, qpairs);
    const int kpairs = Mc * KVc / 2;
    rope_k<<<(kpairs + 255) / 256, 256, 0, stream>>>(kf, cosb, sinb, k_b, KVHc, kpairs);
  }
  transpose_v_k<<<(Bc * KVHc * HDc * Sc + 255) / 256, 256, 0, stream>>>(v_b, vt_b);
  attn_kernel<<<dim3(Sc / 16, Bc * Hc), 32, 0, stream>>>(q_b, k_b, vt_b, attn_b);
  gemm(attn_b, wo_b, qf, nullptr, Mc, Dc, Dc);  // qf = attn @ wo^T
  add_rmsnorm_k<<<Mc, 256, 0, stream>>>(data, qf, fnw, resid, hn_b);

  // --- MoE FFN ---
  gate_k<<<Mc, 256, 0, stream>>>(hn_b, gatew, ew);
  gemm(hn_b, w1_b, c1f, c1b, Mc, Fc, Dc);
  gemm(hn_b, w3_b, c3f, c3b, Mc, Fc, Dc);
  zero_f32_k<<<2048, 256, 0, stream>>>(finf, (size_t)Mc * Dc);

  const size_t nMF = (size_t)Mc * Fc;
  const size_t nMD = (size_t)Mc * Dc;
  for (int e = 0; e < Ec; ++e) {
    gemm16(c1b, w1A_b + (size_t)e * Rc * Fc, t1f, Fc);
    gemm16(c3b, w3A_b + (size_t)e * Rc * Fc, t3f, Fc);
    fused_lora_silu_k<<<(int)((nMF + 255) / 256), 256, 0, stream>>>(
        c1f, c3f, t1f, t3f, w1B + (size_t)e * Fc * Rc, w3B + (size_t)e * Fc * Rc, g_b);
    gemm(g_b, w2_b, qf, hs_b, Mc, Dc, Fc);  // qf = hs fp32, hs_b bf16
    gemm16(hs_b, w2A_b + (size_t)e * Rc * Dc, t2f, Dc);
    lora2_accum_k<<<(int)((nMD + 255) / 256), 256, 0, stream>>>(
        qf, t2f, w2B + (size_t)e * Dc * Rc, ew, e, finf);
  }

  add_out_k<<<(int)((nMD + 255) / 256), 256, 0, stream>>>(resid, finf, out, nMD);
}
